// RGATModel_15728170238295
// MI455X (gfx1250) — matrix operand order repaired and checked by reading, not compiled, not run on hardware
//
#include <hip/hip_runtime.h>
#include <hip/hip_bf16.h>

#define DEVINL __device__ __forceinline__

typedef __attribute__((ext_vector_type(16))) __bf16 v16bf;
typedef __attribute__((ext_vector_type(8)))  __bf16 v8bf;
typedef __attribute__((ext_vector_type(8)))  float  v8f;

constexpr int NH = 8, NB = 32, NN = 512, DIN = 64, DH = 32;
constexpr float NEGFILL = -9.0e15f;
constexpr float SLOPE = 0.2f;
constexpr float LOG2E = 1.4426950408889634f;

// D = A(16x32 bf16) x B(32x16 bf16) + C(16x16 f32)
DEVINL v8f wmma_bf16(v16bf a, v16bf b, v8f c) {
  return __builtin_amdgcn_wmma_f32_16x16x32_bf16(
      /*neg_a=*/false, a, /*neg_b=*/false, b,
      /*c_mod=*/(short)0, c, /*reuse_a=*/false, /*reuse_b=*/false);
}

// A-fragment (16x32, 16-bit) K index for element e, lane-half hl:
//   e=0..15 -> K = 16*hl + e       (contiguous, matches B packing)
DEVINL int a_koff(int e, int hl) { return e + 16 * hl; }

// ds_swizzle BITMASK mode: and_mask=0x1f, or=0, xor=MASK (no bounds guards)
template <int MASK>
DEVINL float swz_xor(float v) {
  return __int_as_float(
      __builtin_amdgcn_ds_swizzle(__float_as_int(v), 0x7C00 | MASK));
}
DEVINL float lane_bcast(float v, int srcLane) {
  return __int_as_float(
      __builtin_amdgcn_ds_bpermute(srcLane << 2, __float_as_int(v)));
}

// Load an A-fragment row (>=32 floats) with 4 x b128 instead of 16 x b32.
DEVINL v16bf load_a_row(const float* __restrict__ row, int hl) {
  const float4* plo = (const float4*)(row + 16 * hl);
  const float4* phi = (const float4*)(row + 16 * hl + 8);
  float4 u0 = plo[0], u1 = plo[1], w0 = phi[0], w1 = phi[1];
  v16bf a;
  a[0] = (__bf16)u0.x; a[1] = (__bf16)u0.y; a[2]  = (__bf16)u0.z; a[3]  = (__bf16)u0.w;
  a[4] = (__bf16)u1.x; a[5] = (__bf16)u1.y; a[6]  = (__bf16)u1.z; a[7]  = (__bf16)u1.w;
  a[8] = (__bf16)w0.x; a[9] = (__bf16)w0.y; a[10] = (__bf16)w0.z; a[11] = (__bf16)w0.w;
  a[12] = (__bf16)w1.x; a[13] = (__bf16)w1.y; a[14] = (__bf16)w1.z; a[15] = (__bf16)w1.w;
  return a;
}

DEVINL float sigm(float x) { return 1.0f / (1.0f + __expf(-x)); }

// ---------------------------------------------------------------------------
// Weight-fragment packing layout in ws (bf16, 512 elems = 1 KB per fragment):
//   fid 0..11 : W_ir/W_ii/W_in  -> fid = wi*4 + k0sel*2 + kc     (wi = 0,1,2)
//   fid 12..17: W_hr/W_hi/W_hm  -> fid = 12 + wh*2 + k0sel       (wh = 0,1,2)
//   fid 18..49: linear per head -> fid = 18 + hd*4 + k0sel*2 + kc
// Within a fragment: lane L's 16 bf16 contiguous at lane*16:
//   elem e = W[(r0 + e + 16*hl)*32 + k0 + nc]   (B-frag: K = e + 16*hl, N = nc)
// ---------------------------------------------------------------------------
__global__ __launch_bounds__(32) void k0_pack(const float* __restrict__ lin,
                                              const float* __restrict__ W_ir,
                                              const float* __restrict__ W_ii,
                                              const float* __restrict__ W_in,
                                              const float* __restrict__ W_hr,
                                              const float* __restrict__ W_hi,
                                              const float* __restrict__ W_hm,
                                              __bf16* __restrict__ wpack) {
  const int fid = blockIdx.x;
  const int lane = threadIdx.x, hl = lane >> 4, nc = lane & 15;
  const float* W;
  int r0, k0;
  if (fid < 12) {
    int wi = fid >> 2;
    W = wi == 0 ? W_ir : (wi == 1 ? W_ii : W_in);
    k0 = ((fid >> 1) & 1) * 16;
    r0 = (fid & 1) * 32;
  } else if (fid < 18) {
    int wh = (fid - 12) >> 1;
    W = wh == 0 ? W_hr : (wh == 1 ? W_hi : W_hm);
    k0 = ((fid - 12) & 1) * 16;
    r0 = 0;
  } else {
    int rel = fid - 18;
    W = lin + (size_t)(rel >> 2) * DIN * DH;
    k0 = ((rel >> 1) & 1) * 16;
    r0 = (rel & 1) * 32;
  }
  __bf16* dst = wpack + (size_t)fid * 512 + lane * 16;
#pragma unroll
  for (int e = 0; e < 16; ++e)
    dst[e] = (__bf16)W[(size_t)(r0 + e + 16 * hl) * DH + k0 + nc];
}

// ---------------------------------------------------------------------------
// Kernel 1: per-head projection + transposed bf16 store + fused e_src/e_tar.
//   HT[hd][b][k][n] (bf16)  -- k-major so k2's B fragments are contiguous.
// One wave per (head, batch, 16-row tile): 4 WMMAs.
// ---------------------------------------------------------------------------
__global__ __launch_bounds__(128) void k1_proj(const float* __restrict__ x,
                                               const __bf16* __restrict__ wpack,
                                               const float* __restrict__ att_src,
                                               const float* __restrict__ att_tar,
                                               __bf16* __restrict__ HT,
                                               float* __restrict__ e_src,
                                               float* __restrict__ e_tar) {
  const int lane = threadIdx.x & 31, wave = threadIdx.x >> 5;
  const int g = blockIdx.x * 4 + wave;          // [0, NH*NB*NN/16)
  const int tpb = NN / 16;
  const int tile = g % tpb, b = (g / tpb) % NB, hd = g / (tpb * NB);
  const int i0 = tile * 16, hl = lane >> 4, nc = lane & 15;

  const float* xr = x + ((size_t)b * NN + i0 + nc) * DIN;   // A row M = nc
  v16bf a0 = load_a_row(xr, hl);
  v16bf a1 = load_a_row(xr + 32, hl);

  const __bf16* fb = wpack + ((size_t)(18 + hd * 4)) * 512 + lane * 16;
  v16bf b00 = *(const v16bf*)(fb + 0 * 512);   // k0=0,  kc=0
  v16bf b01 = *(const v16bf*)(fb + 1 * 512);   // k0=0,  kc=1
  v16bf b10 = *(const v16bf*)(fb + 2 * 512);   // k0=16, kc=0
  v16bf b11 = *(const v16bf*)(fb + 3 * 512);   // k0=16, kc=1

  v8f c0 = {}, c1 = {};
  c0 = wmma_bf16(a0, b00, c0);
  c0 = wmma_bf16(a1, b01, c0);
  c1 = wmma_bf16(a0, b10, c1);
  c1 = wmma_bf16(a1, b11, c1);

  // transposed bf16 store: C vgpr r = row i0+r+8*hl, so 8 rows are contiguous
  __bf16* hb = HT + ((size_t)hd * NB + b) * DH * NN;
  v8bf t0, t1;
#pragma unroll
  for (int r = 0; r < 8; ++r) { t0[r] = (__bf16)c0[r]; t1[r] = (__bf16)c1[r]; }
  *(v8bf*)(hb + (size_t)nc * NN + i0 + 8 * hl)        = t0;
  *(v8bf*)(hb + (size_t)(16 + nc) * NN + i0 + 8 * hl) = t1;

  // fused rank-1 attention vectors: reduce over k (lanes within a half)
  const float as0 = att_src[hd * DH + nc], as1 = att_src[hd * DH + 16 + nc];
  const float at0 = att_tar[hd * DH + nc], at1 = att_tar[hd * DH + 16 + nc];
#pragma unroll
  for (int r = 0; r < 8; ++r) {
    float ps = c0[r] * as0 + c1[r] * as1;
    float pt = c0[r] * at0 + c1[r] * at1;
    ps += swz_xor<1>(ps); ps += swz_xor<2>(ps);
    ps += swz_xor<4>(ps); ps += swz_xor<8>(ps);
    pt += swz_xor<1>(pt); pt += swz_xor<2>(pt);
    pt += swz_xor<4>(pt); pt += swz_xor<8>(pt);
    if (nc == 0) {
      size_t o = ((size_t)hd * NB + b) * NN + i0 + r + 8 * hl;
      e_src[o] = ps;
      e_tar[o] = pt;
    }
  }
}

// ---------------------------------------------------------------------------
// Kernel 2: fused attention + head-mean + tanh(+bias).
// Block = 8 waves = 8 heads, owns a 16-row tile of one batch.
// One-pass softmax: scores are rank-1 and lrelu is monotone, so the exact
// row-max bound is lrelu(e_tar[i] + max_j e_src[j]) -- no online rescale.
// All e-values pre-scaled by log2(e) so exp uses the native v_exp_f32 (2^x).
// ---------------------------------------------------------------------------
__global__ __launch_bounds__(256) void k2_attn(const __bf16* __restrict__ HT,
                                               const float* __restrict__ e_src,
                                               const float* __restrict__ e_tar,
                                               const float* __restrict__ bias,
                                               float* __restrict__ msgs) {
  __shared__ float esh[NH * NN];        // staged e_src*log2e per head, 16 KB
  __shared__ float osh[NH * 16 * DH];   // per-head output tiles,       16 KB
  const int lane = threadIdx.x & 31, hd = threadIdx.x >> 5;
  const int b = blockIdx.x / (NN / 16), i0 = (blockIdx.x % (NN / 16)) * 16;
  const int hl = lane >> 4, nc = lane & 15;

  // stage e_src (scaled) and reduce its max across the wave
  float lm = -1.0e30f;
  {
    const float4* esp4 = (const float4*)(e_src + ((size_t)hd * NB + b) * NN);
    float4* esh4 = (float4*)(esh + hd * NN);
    for (int t = lane; t < NN / 4; t += 32) {
      float4 v = esp4[t];
      v.x *= LOG2E; v.y *= LOG2E; v.z *= LOG2E; v.w *= LOG2E;
      esh4[t] = v;
      lm = fmaxf(lm, fmaxf(fmaxf(v.x, v.y), fmaxf(v.z, v.w)));
    }
  }
  lm = fmaxf(lm, swz_xor<1>(lm));
  lm = fmaxf(lm, swz_xor<2>(lm));
  lm = fmaxf(lm, swz_xor<4>(lm));
  lm = fmaxf(lm, swz_xor<8>(lm));
  lm = fmaxf(lm, swz_xor<16>(lm));
  __syncthreads();

  const float et2 = e_tar[((size_t)hd * NB + b) * NN + i0 + nc] * LOG2E;
  const int gi = i0 + nc;
  const int jdiag = (i0 >> 5) << 5;     // wave-uniform chunk holding diagonal
  const float vm = et2 + lm;
  const float m2 = fmaxf(vm, SLOPE * vm);  // exact row-max bound (scaled)

  const __bf16* hb = HT + ((size_t)hd * NB + b) * DH * NN;
  const __bf16* p0 = hb + (size_t)nc * NN + 16 * hl;        // k = nc
  const __bf16* p1 = hb + (size_t)(16 + nc) * NN + 16 * hl; // k = 16+nc
  const float* ep = esh + hd * NN + 16 * hl;

  v8f c0 = {}, c1 = {};
  float lsum = 0.f;

  for (int j0 = 0; j0 < NN; j0 += 32) {
    if (j0 + 32 < NN) {
      __builtin_prefetch(p0 + j0 + 32, 0, 0);
      __builtin_prefetch(p1 + j0 + 32, 0, 0);
    }
    // vectorized LDS reads of this lane's 16 contiguous K values
    float4 q0 = *(const float4*)(ep + j0);
    float4 q1 = *(const float4*)(ep + j0 + 4);
    float4 q2 = *(const float4*)(ep + j0 + 8);
    float4 q3 = *(const float4*)(ep + j0 + 12);
    float s[16];
    s[0] = et2 + q0.x;  s[1] = et2 + q0.y;  s[2]  = et2 + q0.z;  s[3]  = et2 + q0.w;
    s[4] = et2 + q1.x;  s[5] = et2 + q1.y;  s[6]  = et2 + q1.z;  s[7]  = et2 + q1.w;
    s[8] = et2 + q2.x;  s[9] = et2 + q2.y;  s[10] = et2 + q2.z;  s[11] = et2 + q2.w;
    s[12] = et2 + q3.x; s[13] = et2 + q3.y; s[14] = et2 + q3.z;  s[15] = et2 + q3.w;
#pragma unroll
    for (int e = 0; e < 16; ++e) s[e] = fmaxf(s[e], SLOPE * s[e]);  // lrelu
    if (j0 == jdiag) {                  // uniform branch: diagonal chunk only
#pragma unroll
      for (int e = 0; e < 16; ++e)
        if (j0 + a_koff(e, hl) == gi) s[e] = NEGFILL;
    }
    v16bf pa;
#pragma unroll
    for (int e = 0; e < 16; ++e) {
      float p = __builtin_amdgcn_exp2f(s[e] - m2);
      lsum += p;
      pa[e] = (__bf16)p;
    }
    // B fragments: element e <-> K=j0+e+16*hl, N=nc -> 16 contiguous bf16
    v16bf b0 = *(const v16bf*)(p0 + j0);
    v16bf b1 = *(const v16bf*)(p1 + j0);
    c0 = wmma_bf16(pa, b0, c0);
    c1 = wmma_bf16(pa, b1, c1);
  }

  lsum += swz_xor<16>(lsum);            // full-row denominator
  float invl = 1.0f / lsum;
#pragma unroll
  for (int r = 0; r < 8; ++r) {
    float ar = lane_bcast(invl, r + 8 * hl);
    int row = r + 8 * hl;
    osh[(hd * 16 + row) * DH + nc]      = c0[r] * ar;
    osh[(hd * 16 + row) * DH + 16 + nc] = c1[r] * ar;
  }
  __syncthreads();

  // head-mean + bias + tanh -> msgs[b, i0:i0+16, :]
  for (int idx = threadIdx.x; idx < 16 * DH; idx += 256) {
    float sum = 0.f;
#pragma unroll
    for (int w = 0; w < NH; ++w) sum += osh[w * 16 * DH + idx];
    int k = idx & (DH - 1);
    msgs[((size_t)b * NN + i0 + (idx >> 5)) * DH + k] =
        tanhf(sum * (1.0f / NH) + bias[k]);
  }
}

// ---------------------------------------------------------------------------
// Kernel 3: fused GRU-style gated update. One wave per 16-row tile:
// 12 WMMAs for inputs@W_i* (K=64) + 6 WMMAs for msgs@W_h* (K=32).
// Weight B fragments come pre-packed (contiguous 32B per lane).
// ---------------------------------------------------------------------------
__global__ __launch_bounds__(128) void k3_gru(const float* __restrict__ x,
                                              const float* __restrict__ hid,
                                              const float* __restrict__ msgs,
                                              const __bf16* __restrict__ wpack,
                                              const float* __restrict__ b_ir,
                                              const float* __restrict__ b_ii,
                                              const float* __restrict__ b_in,
                                              float* __restrict__ out) {
  const int lane = threadIdx.x & 31, wave = threadIdx.x >> 5;
  const int g = blockIdx.x * 4 + wave;       // [0, NB*NN/16)
  const int tpb = NN / 16;
  const int b = g / tpb, i0 = (g % tpb) * 16;
  const int hl = lane >> 4, nc = lane & 15;

  const float* xr = x + ((size_t)b * NN + i0 + nc) * DIN;
  const float* mr = msgs + ((size_t)b * NN + i0 + nc) * DH;
  v16bf a0 = load_a_row(xr, hl);
  v16bf a1 = load_a_row(xr + 32, hl);
  v16bf am = load_a_row(mr, hl);

  auto wfrag = [&](int fid) {
    return *(const v16bf*)(wpack + (size_t)fid * 512 + lane * 16);
  };
  auto gin = [&](int wi, int k0s) {          // inputs[16,64] @ W[64,32] tile
    v8f c = {};
    c = wmma_bf16(a0, wfrag(wi * 4 + k0s * 2 + 0), c);
    c = wmma_bf16(a1, wfrag(wi * 4 + k0s * 2 + 1), c);
    return c;
  };
  auto gh = [&](int wh, int k0s) {           // msgs[16,32] @ W[32,32] tile
    v8f c = {};
    return wmma_bf16(am, wfrag(12 + wh * 2 + k0s), c);
  };

  v8f m0, m1;                                // reset gate m  (W_ir, W_hr)
  {
    v8f x0 = gin(0, 0), x1 = gin(0, 1);
    v8f h0 = gh(0, 0), h1 = gh(0, 1);
    float bb0 = b_ir[nc], bb1 = b_ir[16 + nc];
#pragma unroll
    for (int r = 0; r < 8; ++r) {
      m0[r] = sigm(x0[r] + bb0 + h0[r]);
      m1[r] = sigm(x1[r] + bb1 + h1[r]);
    }
  }
  v8f ig0, ig1;                              // input gate i  (W_ii, W_hi)
  {
    v8f x0 = gin(1, 0), x1 = gin(1, 1);
    v8f h0 = gh(1, 0), h1 = gh(1, 1);
    float bb0 = b_ii[nc], bb1 = b_ii[16 + nc];
#pragma unroll
    for (int r = 0; r < 8; ++r) {
      ig0[r] = sigm(x0[r] + bb0 + h0[r]);
      ig1[r] = sigm(x1[r] + bb1 + h1[r]);
    }
  }
  v8f n0, n1;                                // candidate n   (W_in, W_hm)
  {
    v8f x0 = gin(2, 0), x1 = gin(2, 1);
    v8f h0 = gh(2, 0), h1 = gh(2, 1);
    float bb0 = b_in[nc], bb1 = b_in[16 + nc];
#pragma unroll
    for (int r = 0; r < 8; ++r) {
      n0[r] = tanhf(x0[r] + bb0 + m0[r] * h0[r]);
      n1[r] = tanhf(x1[r] + bb1 + m1[r] * h1[r]);
    }
  }
#pragma unroll
  for (int r = 0; r < 8; ++r) {
    int row = i0 + r + 8 * hl;
    size_t o0 = ((size_t)b * NN + row) * DH + nc;
    size_t o1 = o0 + 16;
    float h0v = hid[o0], h1v = hid[o1];
    out[o0] = ig0[r] * n0[r] + (1.0f - ig0[r]) * h0v;
    out[o1] = ig1[r] * n1[r] + (1.0f - ig1[r]) * h1v;
  }
}

// ---------------------------------------------------------------------------
extern "C" void kernel_launch(void* const* d_in, const int* in_sizes, int n_in,
                              void* d_out, int out_size, void* d_ws, size_t ws_size,
                              hipStream_t stream) {
  (void)in_sizes; (void)n_in; (void)out_size; (void)ws_size;
  const float* x     = (const float*)d_in[0];   // inputs  [B,N,64]
  const float* hid   = (const float*)d_in[1];   // hidden  [B,N,32]
  const float* lin   = (const float*)d_in[2];   // linear  [H,64,32]
  const float* bias  = (const float*)d_in[3];   // bias    [1,32]
  const float* att_s = (const float*)d_in[4];   // att_src [H,32]
  const float* att_t = (const float*)d_in[5];   // att_tar [H,32]
  const float* W_hr  = (const float*)d_in[6];
  const float* W_hi  = (const float*)d_in[7];
  const float* W_hm  = (const float*)d_in[8];
  const float* W_ir  = (const float*)d_in[9];
  const float* b_ir  = (const float*)d_in[10];
  const float* W_ii  = (const float*)d_in[11];
  const float* b_ii  = (const float*)d_in[12];
  const float* W_in_ = (const float*)d_in[13];
  const float* b_in_ = (const float*)d_in[14];
  float* out = (float*)d_out;

  char* ws = (char*)d_ws;
  __bf16* HT   = (__bf16*)ws;                                  // 8 MB
  float* e_src = (float*)(ws + (size_t)NH * NB * DH * NN * 2); // 512 KB
  float* e_tar = e_src + (size_t)NH * NB * NN;                 // 512 KB
  float* msgs  = e_tar + (size_t)NH * NB * NN;                 // 2 MB
  __bf16* wpack = (__bf16*)(msgs + (size_t)NB * NN * DH);      // 50 KB

  k0_pack<<<50, 32, 0, stream>>>(lin, W_ir, W_ii, W_in_, W_hr, W_hi, W_hm, wpack);
  k1_proj<<<NH * NB * (NN / 16) / 4, 128, 0, stream>>>(x, wpack, att_s, att_t,
                                                       HT, e_src, e_tar);
  k2_attn<<<NB * (NN / 16), 256, 0, stream>>>(HT, e_src, e_tar, bias, msgs);
  k3_gru<<<NB * (NN / 16) / 4, 128, 0, stream>>>(x, hid, msgs, wpack, b_ir,
                                                 b_ii, b_in_, out);
}
